// KNNModel_64390149701942
// MI455X (gfx1250) — 8-bit matrix operand order repaired and checked by reading, compiled, not run on hardware
//
#include <hip/hip_runtime.h>
#include <hip/hip_bf16.h>

typedef __attribute__((ext_vector_type(8))) int v8i;

#define VOCAB     32
#define UTT_LEN   16
#define NQ        2048
#define KDIM      512
#define SUPPORT   100000
#define STILES    (SUPPORT / 16)
#define KSTEPS    (KDIM / 64)
#define QBLK      256
#define NTILES    (QBLK / 16)
#define GRIDX     32
#define TMEAN     5
#define MPT       10

__global__ void convert_support_f32_to_i8(const float4* __restrict__ sup,
                                          unsigned int* __restrict__ dst, int n4) {
  int i = blockIdx.x * blockDim.x + threadIdx.x;
  if (i >= n4) return;
  float4 f = sup[i];
  unsigned int w = ((unsigned int)f.x)        | ((unsigned int)f.y << 8) |
                   ((unsigned int)f.z << 16)  | ((unsigned int)f.w << 24);
  dst[i] = w;
}

__global__ void build_query_onehot(const int* __restrict__ utts,
                                   unsigned int* __restrict__ Aq,
                                   unsigned int* __restrict__ gkeys) {
  int i = blockIdx.x * blockDim.x + threadIdx.x;
  if (i < NQ) gkeys[i] = 0u;
  if (i >= NQ * (KDIM / 4)) return;
  int n   = i >> 7;
  int d   = i & 127;
  int pos = d >> 3;
  int vo  = (d & 7) << 2;
  int tok = utts[pos * NQ + n];
  int r   = tok - vo;
  Aq[i] = (r >= 0 && r < 4) ? (1u << (r * 8)) : 0u;
}

__global__ void __launch_bounds__(512)
knn_match_wmma(const unsigned char* __restrict__ Aq,
               const unsigned char* __restrict__ supi8,
               unsigned int* __restrict__ gkeys)
{
  extern __shared__ unsigned char smem[];
  unsigned int*  lbest = (unsigned int*)smem;
  unsigned char* lq    = smem + QBLK * sizeof(unsigned int);

  const int tid  = threadIdx.x;
  const int lane = tid & 31;
  const int wave = tid >> 5;
  const int half = lane >> 4;
  const int l15  = lane & 15;
  const int q0   = blockIdx.y * QBLK;

  for (int i = tid; i < QBLK; i += blockDim.x) lbest[i] = 0u;
  {
    const int4* src  = (const int4*)(Aq + (size_t)q0 * KDIM);
    int4*       dstv = (int4*)lq;
    for (int i = tid; i < QBLK * KDIM / 16; i += blockDim.x) dstv[i] = src[i];
  }
  __syncthreads();

  int tilesPerBlk = (STILES + GRIDX - 1) / GRIDX;
  int t0 = blockIdx.x * tilesPerBlk;
  int t1 = t0 + tilesPerBlk; if (t1 > STILES) t1 = STILES;

  for (int st = t0 + wave; st < t1; st += 16) {
    const unsigned char* arow = supi8 + ((size_t)st * 16 + l15) * KDIM + half * 16;
    __builtin_prefetch(arow + (size_t)16 * 16 * KDIM, 0, 0);
    v8i a[KSTEPS];
#pragma unroll
    for (int ks = 0; ks < KSTEPS; ++ks) {
      const int2* p = (const int2*)(arow + ks * 64);
      v8i af;
      af[0] = p[0].x; af[1] = p[0].y;
      af[2] = p[1].x; af[3] = p[1].y;
      af[4] = p[4].x; af[5] = p[4].y;
      af[6] = p[5].x; af[7] = p[5].y;
      a[ks] = af;
    }
    const int sbase = st * 16 + half * 8;

    for (int nt = 0; nt < NTILES; ++nt) {
      const unsigned char* qrow = lq + (size_t)(nt * 16 + l15) * KDIM + half * 16;
      v8i c = {0, 0, 0, 0, 0, 0, 0, 0};
#pragma unroll
      for (int ks = 0; ks < KSTEPS; ++ks) {
        const int4* p = (const int4*)(qrow + ks * 64);
        int4 lo = p[0];
        int4 hi = p[2];
        v8i b = {lo.x, lo.y, lo.z, lo.w, hi.x, hi.y, hi.z, hi.w};
        c = __builtin_amdgcn_wmma_i32_16x16x64_iu8(false, a[ks], false, b, c,
                                                   false, false);
      }
      unsigned int best = 0u;
#pragma unroll
      for (int j = 0; j < 8; ++j) {
        unsigned int key = ((unsigned int)c[j] << 17) |
                           (131071u - (unsigned int)(sbase + j));
        best = key > best ? key : best;
      }
      atomicMax(&lbest[nt * 16 + l15], best);
    }
  }

  __syncthreads();
  for (int i = tid; i < QBLK; i += blockDim.x)
    atomicMax(&gkeys[q0 + i], lbest[i]);
}

__global__ void finalize_meanings(const unsigned int* __restrict__ gkeys,
                                  const int* __restrict__ meanings,
                                  float* __restrict__ out) {
  int n = blockIdx.x * blockDim.x + threadIdx.x;
  if (n >= NQ) return;
  unsigned int key = gkeys[n];
  int s = 131071 - (int)(key & 0x1FFFFu);
#pragma unroll
  for (int t = 0; t < TMEAN; ++t) {
    int m = meanings[s * TMEAN + t];
#pragma unroll
    for (int v = 0; v < MPT; ++v)
      out[n * (TMEAN * MPT) + t * MPT + v] = (v == m) ? 1.0f : 0.0f;
  }
}

extern "C" void kernel_launch(void* const* d_in, const int* in_sizes, int n_in,
                              void* d_out, int out_size, void* d_ws, size_t ws_size,
                              hipStream_t stream) {
  const int*   utts     = (const int*)d_in[0];
  const float* support  = (const float*)d_in[1];
  const int*   meanings = (const int*)d_in[2];
  float*       out      = (float*)d_out;

  unsigned char* supi8 = (unsigned char*)d_ws;
  unsigned char* Aq    = supi8 + (size_t)SUPPORT * KDIM;
  unsigned int*  gkeys = (unsigned int*)(Aq + (size_t)NQ * KDIM);

  int n4 = SUPPORT * KDIM / 4;
  convert_support_f32_to_i8<<<(n4 + 255) / 256, 256, 0, stream>>>(
      (const float4*)support, (unsigned int*)supi8, n4);

  int nq4 = NQ * KDIM / 4;
  build_query_onehot<<<(nq4 + 255) / 256, 256, 0, stream>>>(
      utts, (unsigned int*)Aq, gkeys);

  size_t smem = QBLK * sizeof(unsigned int) + (size_t)QBLK * KDIM;
  hipFuncSetAttribute((const void*)knn_match_wmma,
                      hipFuncAttributeMaxDynamicSharedMemorySize, (int)smem);
  dim3 grid(GRIDX, NQ / QBLK);
  knn_match_wmma<<<grid, 512, smem, stream>>>(Aq, supi8, gkeys);

  finalize_meanings<<<(NQ + 255) / 256, 256, 0, stream>>>(gkeys, meanings, out);
}
